// RNAFoldingModel_76115410420411
// MI455X (gfx1250) — matrix operand order repaired and checked by reading, not compiled, not run on hardware
//
#include <hip/hip_runtime.h>
#include <hip/hip_bf16.h>

typedef __bf16 bf16_t;
typedef __attribute__((ext_vector_type(16))) bf16_t v16bf;
typedef __attribute__((ext_vector_type(8)))  bf16_t v8bf;
typedef __attribute__((ext_vector_type(8)))  float  v8f;
typedef __attribute__((ext_vector_type(4)))  unsigned int v4u;
typedef __attribute__((ext_vector_type(8)))  int v8i_;
typedef __attribute__((ext_vector_type(4)))  int v4i_;

#define B_ 8
#define S_ 512
#define E_ 512
#define H_ 8
#define D_ 64
#define K_ 512   // GEMM inner dim (= E)
#define M_ 4096  // B*S

#if defined(__has_builtin)
#if __has_builtin(__builtin_amdgcn_tensor_load_to_lds) && __has_builtin(__builtin_amdgcn_s_wait_tensorcnt)
#define HAVE_TDM 1
#endif
#endif
#ifndef HAVE_TDM
#define HAVE_TDM 0
#endif

// ---------------------------------------------------------------------------
// A-fragment (16-bit, 16x32): lane L holds row M=L&15.
//   lanes 0-15 : halves 0..15 = K b0..b0+15, b0=0
//   lanes 16-31: same with b0=16
// -> two contiguous 16B chunks at (k0+b0) and (k0+b0+8).
static __device__ __forceinline__ v16bf load_a_frag(const bf16_t* p) {
    v8bf lo = *(const v8bf*)p;
    v8bf hi = *(const v8bf*)(p + 8);
    v16bf a;
#pragma unroll
    for (int t = 0; t < 8; ++t) { a[t] = lo[t]; a[t + 8] = hi[t]; }
    return a;
}

// ---------------------------------------------------------------------------
// fp32 -> bf16 elementwise convert
__global__ void cvt_kernel(const float* __restrict__ in, bf16_t* __restrict__ out, int n) {
    int i = blockIdx.x * blockDim.x + threadIdx.x;
    if (i < n) out[i] = (bf16_t)in[i];
}

// ---------------------------------------------------------------------------
// Y[m,n] = sum_k A[m,k] * W[n,k] + bias[n]   (A,W bf16 row-major; WMMA bf16)
// 32x64 tile per wave (2 A-frags x 4 B-frags -> 8 WMMA per k-step).
// mode 0: store bf16 q  [b,h,s,d], scaled by 1/8 (1/sqrt(D))
// mode 1: store bf16 k  [b,h,s,d]
// mode 2: store bf16 vT [b,h,d,s]
// mode 3: store fp32 to outF[m*E_+n]
__global__ __launch_bounds__(128) void gemm_bf16(const bf16_t* __restrict__ A,
                                                 const bf16_t* __restrict__ W,
                                                 const float*  __restrict__ bias,
                                                 float* outF, bf16_t* outB, int mode) {
#if HAVE_TDM
    __shared__ bf16_t stage[32 * K_];   // 32 KB TDM prefetch target
#endif
    const int lane = threadIdx.x & 31;
    const int wave = threadIdx.x >> 5;
    const int m0 = blockIdx.x * 32;
    const int n0 = blockIdx.y * 256 + wave * 64;

#if HAVE_TDM
    // TDM prefetch of this block's 32x512 bf16 A-tile into LDS (wave 0 only).
    // Used as an async L2/LDS prefetch; compute reads the normal global path.
    if (wave == 0) {
        unsigned long long ga = (unsigned long long)(uintptr_t)(A + (size_t)m0 * K_);
        unsigned int lds_off = (unsigned int)(uintptr_t)&stage[0];
        v4u g0 = { 1u,                    // count=1 (valid descriptor)
                   lds_off,               // lds_addr
                   (unsigned int)ga,      // global_addr[31:0]
                   (unsigned int)((ga >> 32) & 0x01FFFFFFu) | (2u << 30) }; // addr[56:32] | type=2
        v8i_ g1 = { 0x00010000,           // workgroup_mask=0, data_size=1 (2B)
                    (int)(512u << 16),    // tensor_dim0 = 512 (bits 63:48)
                    (int)(4096u << 16),   // tensor_dim1 = 4096 (bits 95:80)
                    (int)(512u << 16),    // tile_dim0 = 512 (bits 127:112)
                    32,                   // tile_dim1 = 32
                    512,                  // tensor_dim0_stride = 512
                    0, 0 };
        v4i_ gz4 = { 0, 0, 0, 0 };
        v8i_ gz8 = { 0, 0, 0, 0, 0, 0, 0, 0 };
        __builtin_amdgcn_tensor_load_to_lds(g0, g1, gz4, gz4, gz8, 0);
    }
#endif

    const int b0    = (lane >= 16) ? 16 : 0;  // A-frag K base
    const int kbase = (lane < 16) ? 0 : 16;   // B-frag K base
    const int nlane = lane & 15;

    v8f acc[2][4];
#pragma unroll
    for (int sub = 0; sub < 2; ++sub)
#pragma unroll
        for (int nt = 0; nt < 4; ++nt)
#pragma unroll
            for (int r = 0; r < 8; ++r) acc[sub][nt][r] = 0.0f;

    for (int k0 = 0; k0 < K_; k0 += 32) {
        v16bf a[2];
#pragma unroll
        for (int sub = 0; sub < 2; ++sub)
            a[sub] = load_a_frag(A + (size_t)(m0 + sub * 16 + nlane) * K_ + k0 + b0);
#pragma unroll
        for (int nt = 0; nt < 4; ++nt) {
            const int n = n0 + nt * 16 + nlane;
            v16bf bm = *(const v16bf*)(W + (size_t)n * K_ + k0 + kbase);
#pragma unroll
            for (int sub = 0; sub < 2; ++sub)
                acc[sub][nt] = __builtin_amdgcn_wmma_f32_16x16x32_bf16(
                    false, a[sub], false, bm, (short)0, acc[sub][nt], false, false);
        }
    }

    const int mrow = (lane >= 16) ? 8 : 0;    // D-matrix row base for this lane
#pragma unroll
    for (int sub = 0; sub < 2; ++sub)
#pragma unroll
        for (int nt = 0; nt < 4; ++nt) {
            const int n  = n0 + nt * 16 + nlane;
            const float bv = bias[n];
            const int h = n >> 6, d = n & 63;
#pragma unroll
            for (int r = 0; r < 8; ++r) {
                const int m  = m0 + sub * 16 + mrow + r;
                const int bb = m >> 9, s = m & 511;
                float val = acc[sub][nt][r] + bv;
                if (mode == 0) {
                    outB[(((size_t)bb * H_ + h) * S_ + s) * D_ + d] = (bf16_t)(val * 0.125f);
                } else if (mode == 1) {
                    outB[(((size_t)bb * H_ + h) * S_ + s) * D_ + d] = (bf16_t)val;
                } else if (mode == 2) {
                    outB[(((size_t)bb * H_ + h) * D_ + d) * S_ + s] = (bf16_t)val;
                } else {
                    outF[(size_t)m * E_ + n] = val;
                }
            }
        }

#if HAVE_TDM
    if (wave == 0) __builtin_amdgcn_s_wait_tensorcnt(0);
#endif
}

// ---------------------------------------------------------------------------
// relws[b,h,i,j] = dot(q[b,h,i,:], rel[i,j,h,:])  -- single pass over rel,
// amortized over all 8 batches. q (bf16, prescaled by 1/8) is rescaled by 8.
__global__ __launch_bounds__(128) void rel_kernel(const float* __restrict__ rel,
                                                  const bf16_t* __restrict__ qb,
                                                  float* __restrict__ relws) {
    const int j = blockIdx.x * 128 + threadIdx.x;
    const int i = blockIdx.y;
    const int h = blockIdx.z;

    __shared__ float qs[B_][D_];
    for (int t = threadIdx.x; t < B_ * D_; t += 128) {
        const int bb = t >> 6, d = t & 63;
        qs[bb][d] = 8.0f * (float)qb[(((size_t)bb * H_ + h) * S_ + i) * D_ + d];
    }
    __syncthreads();

    const float4* rp = (const float4*)(rel + (((size_t)i * S_ + j) * H_ + h) * (size_t)D_);
    __builtin_prefetch((const char*)rp + (size_t)128 * H_ * D_ * 4, 0, 1);

    float acc[B_];
#pragma unroll
    for (int b = 0; b < B_; ++b) acc[b] = 0.0f;

#pragma unroll 4
    for (int d4 = 0; d4 < D_ / 4; ++d4) {
        const float4 rv = rp[d4];
#pragma unroll
        for (int b = 0; b < B_; ++b) {
            acc[b] += rv.x * qs[b][4 * d4 + 0] + rv.y * qs[b][4 * d4 + 1]
                    + rv.z * qs[b][4 * d4 + 2] + rv.w * qs[b][4 * d4 + 3];
        }
    }
#pragma unroll
    for (int b = 0; b < B_; ++b)
        relws[(((size_t)b * H_ + h) * S_ + i) * S_ + j] = acc[b];
}

// ---------------------------------------------------------------------------
// One wave (32 threads) per (b,h,16-row i-tile): scores (WMMA, rel bias in C),
// mask, row softmax in LDS, P->bf16, PV (WMMA), write attn out bf16 [b,s,e].
__global__ __launch_bounds__(32) void attn_kernel(const bf16_t* __restrict__ qb,
                                                  const bf16_t* __restrict__ kb,
                                                  const bf16_t* __restrict__ vtb,
                                                  const float*  __restrict__ relws,
                                                  const unsigned char* __restrict__ mask,
                                                  bf16_t* __restrict__ ao) {
    __shared__ float  sc[16 * S_];   // 32 KB: fp32 score rows
    __shared__ bf16_t pb[16 * S_];   // 16 KB: bf16 probability rows

    const int lane = threadIdx.x;
    const int tile = blockIdx.x;
    const int it = tile & 31, h = (tile >> 5) & 7, b = tile >> 8;
    const int i0 = it * 16;
    const size_t bh = (size_t)b * H_ + h;

    const int b0    = (lane >= 16) ? 16 : 0;
    const int kbase = (lane < 16) ? 0 : 16;
    const int nlane = lane & 15;
    const int mrow  = (lane >= 16) ? 8 : 0;

    // Q A-fragments for full K=64 (two 32-wide k-steps), q prescaled by 1/8
    v16bf qa[2];
#pragma unroll
    for (int ks = 0; ks < 2; ++ks)
        qa[ks] = load_a_frag(qb + (bh * S_ + i0 + nlane) * D_ + ks * 32 + b0);

    // ---- scores = (q/8)·k + rel, masked -> LDS
    for (int jt = 0; jt < 32; ++jt) {
        const int jc = jt * 16 + nlane;
        v8f c;
        const float* rp = relws + (bh * S_ + i0 + mrow) * (size_t)S_ + jc;
#pragma unroll
        for (int r = 0; r < 8; ++r) c[r] = rp[(size_t)r * S_];
#pragma unroll
        for (int ks = 0; ks < 2; ++ks) {
            v16bf bm = *(const v16bf*)(kb + (bh * S_ + jc) * D_ + ks * 32 + kbase);
            c = __builtin_amdgcn_wmma_f32_16x16x32_bf16(
                false, qa[ks], false, bm, (short)0, c, false, false);
        }
        const bool mk = mask[b * S_ + jc] != 0;
#pragma unroll
        for (int r = 0; r < 8; ++r)
            sc[(mrow + r) * S_ + jc] = mk ? -1e9f : c[r];
    }

    // ---- softmax: lane L handles row L&15, half (L>>4); combine via shfl_xor(16)
    const int row = lane & 15;
    const int cb  = (lane >> 4) * 256;
    float mx = -3.4e38f;
    for (int t = 0; t < 256; ++t) mx = fmaxf(mx, sc[row * S_ + cb + t]);
    mx = fmaxf(mx, __shfl_xor(mx, 16));
    float sum = 0.0f;
    for (int t = 0; t < 256; ++t) {
        const float e = __expf(sc[row * S_ + cb + t] - mx);
        sc[row * S_ + cb + t] = e;
        sum += e;
    }
    sum += __shfl_xor(sum, 16);
    const float inv = 1.0f / sum;
    for (int t = 0; t < 256; ++t)
        pb[row * S_ + cb + t] = (bf16_t)(sc[row * S_ + cb + t] * inv);
    // single wave: DS ops are in-order within a wave, no barrier needed

    // ---- out[i,d] = sum_j P[i,j] * V[j,d]  (V stored transposed [b,h,d,s])
    v8f acc[4];
#pragma unroll
    for (int dt = 0; dt < 4; ++dt)
#pragma unroll
        for (int r = 0; r < 8; ++r) acc[dt][r] = 0.0f;

    for (int kt = 0; kt < 16; ++kt) {
        v16bf a = load_a_frag(pb + (size_t)nlane * S_ + kt * 32 + b0);
#pragma unroll
        for (int dt = 0; dt < 4; ++dt) {
            const int d = dt * 16 + nlane;
            v16bf bm = *(const v16bf*)(vtb + (bh * D_ + d) * (size_t)S_ + kt * 32 + kbase);
            acc[dt] = __builtin_amdgcn_wmma_f32_16x16x32_bf16(
                false, a, false, bm, (short)0, acc[dt], false, false);
        }
    }
#pragma unroll
    for (int dt = 0; dt < 4; ++dt)
#pragma unroll
        for (int r = 0; r < 8; ++r) {
            const int srow = i0 + mrow + r;
            ao[((size_t)b * S_ + srow) * E_ + h * D_ + dt * 16 + nlane] = (bf16_t)acc[dt][r];
        }
}

// ---------------------------------------------------------------------------
extern "C" void kernel_launch(void* const* d_in, const int* in_sizes, int n_in,
                              void* d_out, int out_size, void* d_ws, size_t ws_size,
                              hipStream_t stream) {
    const float* query = (const float*)d_in[0];
    const float* key   = (const float*)d_in[1];
    const float* value = (const float*)d_in[2];
    const unsigned char* mask = (const unsigned char*)d_in[3];
    const float* rel = (const float*)d_in[4];
    const float* Wq = (const float*)d_in[5];  const float* bq = (const float*)d_in[6];
    const float* Wk = (const float*)d_in[7];  const float* bk = (const float*)d_in[8];
    const float* Wv = (const float*)d_in[9];  const float* bv = (const float*)d_in[10];
    const float* Wo = (const float*)d_in[11]; const float* bo = (const float*)d_in[12];
    float* out = (float*)d_out;

    const size_t MB = 1024 * 1024;
    char* ws = (char*)d_ws;
    bf16_t* xq  = (bf16_t*)(ws + 0 * MB);    // B*S*E bf16   (4 MB)
    bf16_t* xk  = (bf16_t*)(ws + 4 * MB);
    bf16_t* xv  = (bf16_t*)(ws + 8 * MB);
    bf16_t* wqb = (bf16_t*)(ws + 12 * MB);   // E*E bf16     (0.5 MB)
    bf16_t* wkb = (bf16_t*)(ws + 12 * MB + 512 * 1024);
    bf16_t* wvb = (bf16_t*)(ws + 13 * MB);
    bf16_t* wob = (bf16_t*)(ws + 13 * MB + 512 * 1024);
    bf16_t* qbw = (bf16_t*)(ws + 14 * MB);   // [B,H,S,D] bf16, prescaled 1/8
    bf16_t* kbw = (bf16_t*)(ws + 18 * MB);   // [B,H,S,D] bf16
    bf16_t* vtw = (bf16_t*)(ws + 22 * MB);   // [B,H,D,S] bf16 (transposed)
    bf16_t* aow = (bf16_t*)(ws + 26 * MB);   // [B,S,E]  bf16
    float*  rlw = (float*)(ws + 30 * MB);    // [B,H,S,S] fp32 (64 MB)

    const int nIn = B_ * S_ * E_;   // 2,097,152
    const int nW  = E_ * E_;        //   262,144
    cvt_kernel<<<(nIn + 255) / 256, 256, 0, stream>>>(query, xq, nIn);
    cvt_kernel<<<(nIn + 255) / 256, 256, 0, stream>>>(key,   xk, nIn);
    cvt_kernel<<<(nIn + 255) / 256, 256, 0, stream>>>(value, xv, nIn);
    cvt_kernel<<<(nW + 255) / 256, 256, 0, stream>>>(Wq, wqb, nW);
    cvt_kernel<<<(nW + 255) / 256, 256, 0, stream>>>(Wk, wkb, nW);
    cvt_kernel<<<(nW + 255) / 256, 256, 0, stream>>>(Wv, wvb, nW);
    cvt_kernel<<<(nW + 255) / 256, 256, 0, stream>>>(Wo, wob, nW);

    dim3 ggrid(M_ / 32, E_ / 256);
    gemm_bf16<<<ggrid, 128, 0, stream>>>(xq, wqb, bq, nullptr, qbw, 0);
    gemm_bf16<<<ggrid, 128, 0, stream>>>(xk, wkb, bk, nullptr, kbw, 1);
    gemm_bf16<<<ggrid, 128, 0, stream>>>(xv, wvb, bv, nullptr, vtw, 2);

    rel_kernel<<<dim3(S_ / 128, S_, H_), 128, 0, stream>>>(rel, qbw, rlw);

    attn_kernel<<<B_ * H_ * (S_ / 16), 32, 0, stream>>>(qbw, kbw, vtw, rlw, mask, aow);

    gemm_bf16<<<ggrid, 128, 0, stream>>>(aow, wob, bo, out, nullptr, 3);
}
